// ConvContract_66348654788713
// MI455X (gfx1250) — matrix operand order repaired and checked by reading, not compiled, not run on hardware
//
#include <hip/hip_runtime.h>

// ---------------------------------------------------------------------------
// CDNA5 (gfx1250) implicit-GEMM torus conv, bf16 WMMA (v_wmma_f32_16x16x32_bf16)
//
// Combined view: out[192ch][256][256] = conv3x3_wrap( X[192ch][256][256], W[192][192][3][3] )
//   in  ch:  [0,64)  = x0,   [64,192) = x1 folded (i*2+d)
//   out ch:  [0,64)  = out0, [64,192) = out1 folded (o*2+d')
// GEMM: M = out-ch (192), N = pixels, K = 192*9 = 1728 (chunks of 32 for bf16 WMMA)
// LDS staging uses GLOBAL_LOAD_ASYNC_TO_LDS_B128 (ASYNCcnt) when the builtin
// exists; falls back to vector load/store copy otherwise.
// ---------------------------------------------------------------------------

typedef __attribute__((ext_vector_type(16))) __bf16 v16bf;
typedef __attribute__((ext_vector_type(8)))  __bf16 v8bf;
typedef __attribute__((ext_vector_type(8)))  float  v8f;
typedef __attribute__((ext_vector_type(4)))  int    v4i;

#define THREADS 256
#define XPC 192      // combined input channels
#define XPW 258      // padded width/height (256 + 1 halo each side, torus-wrapped)
#define XPL 200      // LDS row pitch in halfs (pad 192->200: stride 100 dwords, 16 distinct banks)
#define NPIX 65536   // 256*256

#define AW_HALFS  331776              // 9 taps * 12 ocTiles * 6 kChunks * 512
#define XP_HALFS  (XPW*XPW*XPC)       // 12,780,288
#define WS_AW_OFF 0
#define WS_XP_OFF 663552              // AW_HALFS*2
#define WS_MEAN_OFF 26224128          // WS_XP_OFF + XP_HALFS*2

#if defined(__has_builtin)
#if __has_builtin(__builtin_amdgcn_global_load_async_to_lds_b128)
#define HAVE_ASYNC_LDS 1
#endif
#endif

__device__ __forceinline__ unsigned short f2bf(float f) {
  unsigned int u = __float_as_uint(f);
  unsigned int r = u + 0x7FFFu + ((u >> 16) & 1u);   // round-to-nearest-even
  return (unsigned short)(r >> 16);
}

// Assemble a v16bf from two 16B-aligned 8-half chunks (lowers to 2x b128 loads,
// works for both LDS and global pointers; avoids 32B-alignment assumptions).
__device__ __forceinline__ v16bf load16(const unsigned short* p) {
  v8bf lo = *(const v8bf*)(p);
  v8bf hi = *(const v8bf*)(p + 8);
  return __builtin_shufflevector(lo, hi, 0,1,2,3,4,5,6,7,8,9,10,11,12,13,14,15);
}

// ---------------------------------------------------------------------------
// Kernel 1: synthesize combined filter, store bf16 in A-fragment-ready layout.
// Flat index: ((t*12 + mt)*6 + kc)*512 + lane*16 + e
//   A 16x32 bf16 k order used here (contiguous, B matches): lane l (m = l&15, hi = l>>4) holds
//   e in [0,16) -> K = hi*16 + e
// ---------------------------------------------------------------------------
__global__ __launch_bounds__(THREADS) void pack_weights(
    const float* __restrict__ fil0, const float* __restrict__ fil1,
    const float* __restrict__ fil2, const float* __restrict__ w00,
    const float* __restrict__ w01,  const float* __restrict__ w10,
    const float* __restrict__ w11,  unsigned short* __restrict__ AW) {
  int tid = blockIdx.x * THREADS + threadIdx.x;
  if (tid >= AW_HALFS) return;
  int e  = tid & 15;
  int l  = (tid >> 4) & 31;
  int r  = tid >> 9;            // (t*12+mt)*6+kc
  int kc = r % 6;
  int mt = (r / 6) % 12;
  int t  = r / 72;
  int ky = t / 3, kx = t % 3;
  int hi = l >> 4, m = l & 15;
  int oc = mt * 16 + m;
  int kk = hi * 16 + e;
  int ic = kc * 32 + kk;

  float acc = 0.f;
  if (oc < 64) {
    if (ic < 64) {
      for (int k = 0; k < 3; ++k)
        acc += w00[(oc * 64 + ic) * 3 + k] * fil0[(k * 3 + ky) * 3 + kx];
    } else {
      int j = ic - 64, i = j >> 1, d = j & 1;
      for (int k = 0; k < 6; ++k)
        acc += w10[(oc * 64 + i) * 6 + k] * fil1[((k * 3 + ky) * 3 + kx) * 2 + d];
    }
  } else {
    int jo = oc - 64, o = jo >> 1, dd = jo & 1;
    if (ic < 64) {
      for (int k = 0; k < 6; ++k)
        acc += w01[(o * 64 + ic) * 6 + k] * fil1[((k * 3 + ky) * 3 + kx) * 2 + dd];
    } else {
      int j = ic - 64, i = j >> 1, d = j & 1;
      for (int k = 0; k < 12; ++k)
        acc += w11[(o * 64 + i) * 12 + k] *
               fil2[(((k * 3 + ky) * 3 + kx) * 2 + d) * 2 + dd];
    }
  }
  AW[tid] = f2bf(acc);
}

// ---------------------------------------------------------------------------
// Kernel 2: pack inputs to channels-last, torus-padded bf16: Xp[y][x][ic]
// ---------------------------------------------------------------------------
__global__ __launch_bounds__(THREADS) void pack_input(
    const float* __restrict__ x0, const float* __restrict__ x1,
    unsigned short* __restrict__ Xp) {
  int tid = blockIdx.x * THREADS + threadIdx.x;
  if (tid >= XP_HALFS) return;
  int ic = tid % XPC;
  int rest = tid / XPC;
  int x  = rest % XPW;
  int yy = rest / XPW;
  int sy = (yy + 255) & 255;   // padded coord p -> source (p-1) mod 256
  int sx = (x + 255) & 255;
  float v;
  if (ic < 64) {
    v = x0[((size_t)ic * 256 + sy) * 256 + sx];
  } else {
    int j = ic - 64, i = j >> 1, d = j & 1;
    v = x1[(((size_t)i * 256 + sy) * 256 + sx) * 2 + d];
  }
  Xp[tid] = f2bf(v);
}

// ---------------------------------------------------------------------------
// Kernel 3: main WMMA conv. Block = (ocBlock mb of 64) x (image row y, 256 px).
// LDS: the 3 padded rows needed (one contiguous 290KB chunk of Xp), pitch XPL.
// 8 waves: (wid&1) -> 32-oc half, (wid>>1) -> 64-px quarter; 8 accumulators.
// 432 v_wmma per wave.
// ---------------------------------------------------------------------------
__global__ __launch_bounds__(THREADS) void conv_main(
    const unsigned short* __restrict__ AW, const unsigned short* __restrict__ Xp,
    const float* __restrict__ b0, float* __restrict__ out) {
  extern __shared__ unsigned short tile[];   // [3][XPW][XPL]
  const int y   = blockIdx.y;
  const int mb  = blockIdx.x;
  const int tid = threadIdx.x;

  // Stage rows [y, y+3) of Xp (contiguous) into LDS with pitch padding.
  {
    const int nvec = 3 * XPW * (XPC / 8);    // 18576 x 16B chunks
#if defined(HAVE_ASYNC_LDS)
    // gfx1250 async DMA path: global -> LDS with no VGPR payload (ASYNCcnt).
    const char* srcb = (const char*)(Xp + (size_t)y * (XPW * XPC));
    for (int i = tid; i < nvec; i += THREADS) {
      int pair = i / (XPC / 8);              // row*XPW + x
      int q    = i % (XPC / 8);
      __builtin_amdgcn_global_load_async_to_lds_b128(
          (__attribute__((address_space(1))) v4i*)(srcb + (size_t)i * 16),
          (__attribute__((address_space(3))) v4i*)(tile + pair * XPL + q * 8),
          0, 0);
    }
#if __has_builtin(__builtin_amdgcn_s_wait_asynccnt)
    __builtin_amdgcn_s_wait_asynccnt(0);
#else
    asm volatile("s_wait_asynccnt 0x0" ::: "memory");
#endif
#else
    const uint4* src = (const uint4*)(Xp + (size_t)y * (XPW * XPC));
    for (int i = tid; i < nvec; i += THREADS) {
      int pair = i / (XPC / 8);              // row*XPW + x
      int q    = i % (XPC / 8);
      *(uint4*)(tile + pair * XPL + q * 8) = src[i];
    }
#endif
  }
  __syncthreads();

  const int lane = tid & 31;
  const int wid  = tid >> 5;
  const int hi   = lane >> 4;
  const int ln   = lane & 15;
  const int ocW  = mb * 64 + (wid & 1) * 32;   // wave's oc base (2 x 16 tiles)
  const int pxW  = (wid >> 1) * 64;            // wave's pixel base (4 x 16 tiles)

  v8f acc[2][4] = {};

  for (int t = 0; t < 9; ++t) {
    const int ky = t / 3, kx = t - ky * 3;
    const unsigned short* arow =
        AW + (((size_t)t * 12 + (ocW >> 4)) * 6) * 512 + lane * 16;
    if (t < 8)  // gfx1250 global_prefetch_b8 on next tap's weight block
      __builtin_prefetch(AW + (((size_t)(t + 1) * 12 + (ocW >> 4)) * 6) * 512 + lane * 16, 0, 1);
    const unsigned short* brow = tile + (ky * XPW + kx + pxW + ln) * XPL + hi * 16;
#pragma unroll
    for (int kc = 0; kc < 6; ++kc) {
      v16bf a0 = load16(arow + kc * 512);             // oc tile 0 (16x32)
      v16bf a1 = load16(arow + 6 * 512 + kc * 512);   // oc tile 1
#pragma unroll
      for (int nt = 0; nt < 4; ++nt) {
        v16bf b = load16(brow + nt * 16 * XPL + kc * 32);  // 32k x 16px
        acc[0][nt] = __builtin_amdgcn_wmma_f32_16x16x32_bf16(
            false, a0, false, b, (short)0, acc[0][nt], false, false);
        acc[1][nt] = __builtin_amdgcn_wmma_f32_16x16x32_bf16(
            false, a1, false, b, (short)0, acc[1][nt], false, false);
      }
    }
  }

  // C layout: VGPR r -> oc = base + r + hi*8 ; N lane -> px = base + (lane&15)
  const int pxL = pxW + ln;
  if (mb == 0) {
#pragma unroll
    for (int mt = 0; mt < 2; ++mt)
#pragma unroll
      for (int nt = 0; nt < 4; ++nt)
#pragma unroll
        for (int r = 0; r < 8; ++r) {
          int oc = ocW + mt * 16 + hi * 8 + r;
          out[(size_t)oc * NPIX + y * 256 + pxL + nt * 16] =
              acc[mt][nt][r] + b0[oc];
        }
  } else {
    float* out1 = out + (size_t)64 * NPIX;
#pragma unroll
    for (int mt = 0; mt < 2; ++mt)
#pragma unroll
      for (int nt = 0; nt < 4; ++nt)
#pragma unroll
        for (int r = 0; r < 8; ++r) {
          int j = ocW - 64 + mt * 16 + hi * 8 + r;   // folded out1 channel
          out1[((size_t)(j >> 1) * NPIX + y * 256 + pxL + nt * 16) * 2 + (j & 1)] =
              acc[mt][nt][r];
        }
  }
}

// ---------------------------------------------------------------------------
// Kernel 4: deterministic spatial mean of raw out1 per (o, d') channel pair.
// ---------------------------------------------------------------------------
__global__ __launch_bounds__(THREADS) void reduce_mean(
    const float* __restrict__ out1, float* __restrict__ means) {
  __shared__ float red[THREADS];
  const int c = blockIdx.x;           // 0..127 : o = c>>1, d' = c&1
  const float* base = out1 + (size_t)(c >> 1) * (2 * NPIX) + (c & 1);
  float s = 0.f;
  for (int p = threadIdx.x; p < NPIX; p += THREADS) s += base[(size_t)p * 2];
  red[threadIdx.x] = s;
  __syncthreads();
  for (int st = THREADS / 2; st > 0; st >>= 1) {
    if ((int)threadIdx.x < st) red[threadIdx.x] += red[threadIdx.x + st];
    __syncthreads();
  }
  if (threadIdx.x == 0) means[c] = red[0] * (1.0f / (float)NPIX);
}

// ---------------------------------------------------------------------------
// Kernel 5: out1 += mean[o][d'] * b1[o]   (out1 was freshly rewritten above)
// ---------------------------------------------------------------------------
__global__ __launch_bounds__(THREADS) void apply_mean(
    float* __restrict__ out1, const float* __restrict__ means,
    const float* __restrict__ b1) {
  size_t e = (size_t)blockIdx.x * THREADS + threadIdx.x;
  if (e >= (size_t)8388608) return;
  int o  = (int)(e >> 17);
  int dd = (int)(e & 1);
  out1[e] += means[o * 2 + dd] * b1[o];
}

// ---------------------------------------------------------------------------
extern "C" void kernel_launch(void* const* d_in, const int* in_sizes, int n_in,
                              void* d_out, int out_size, void* d_ws, size_t ws_size,
                              hipStream_t stream) {
  const float* x0   = (const float*)d_in[0];
  const float* x1   = (const float*)d_in[1];
  const float* fil0 = (const float*)d_in[2];
  const float* fil1 = (const float*)d_in[3];
  const float* fil2 = (const float*)d_in[4];
  const float* w00  = (const float*)d_in[5];
  const float* w01  = (const float*)d_in[6];
  const float* w10  = (const float*)d_in[7];
  const float* w11  = (const float*)d_in[8];
  const float* b0   = (const float*)d_in[9];
  const float* b1   = (const float*)d_in[10];
  float* out = (float*)d_out;

  unsigned short* AW = (unsigned short*)((char*)d_ws + WS_AW_OFF);
  unsigned short* Xp = (unsigned short*)((char*)d_ws + WS_XP_OFF);
  float* means       = (float*)((char*)d_ws + WS_MEAN_OFF);

  pack_weights<<<AW_HALFS / THREADS, THREADS, 0, stream>>>(fil0, fil1, fil2, w00,
                                                           w01, w10, w11, AW);
  pack_input<<<XP_HALFS / THREADS, THREADS, 0, stream>>>(x0, x1, Xp);

  const size_t smem = (size_t)3 * XPW * XPL * sizeof(unsigned short);  // 309,600 B
  conv_main<<<dim3(3, 256), THREADS, smem, stream>>>(AW, Xp, b0, out);

  reduce_mean<<<128, THREADS, 0, stream>>>(out + (size_t)64 * NPIX, means);
  apply_mean<<<8388608 / THREADS, THREADS, 0, stream>>>(out + (size_t)64 * NPIX,
                                                        means, b1);
  (void)in_sizes; (void)n_in; (void)out_size; (void)ws_size;
}
